// GRU_Decoder_19078244729391
// MI455X (gfx1250) — compile-verified, matrix operand order checked by reading, not run on hardware
//
#include <hip/hip_runtime.h>

// ---------------------------------------------------------------------------
// 2-layer GRU decoder for MI455X (gfx1250, wave32, WMMA bf16 -> f32)
// ---------------------------------------------------------------------------
namespace {
constexpr int B   = 64;
constexpr int T   = 512;
constexpr int IN  = 512;
constexpr int H   = 1024;
constexpr int H3  = 3 * H;
constexpr int K0  = IN + H;   // packed K for layer 0 weights [Wih0 | Whh0] = 1536
constexpr int K1  = H + H;    // packed K for layer 1 weights [Wih1 | Whh1] = 2048
constexpr int NBLK = 32;      // persistent grid
constexpr int NTHR = 256;     // 8 waves per block -> 256 waves total
}

typedef __attribute__((ext_vector_type(16))) __bf16 v16bf;
typedef __attribute__((ext_vector_type(8)))  __bf16 v8bf;
typedef __attribute__((ext_vector_type(8)))  float  v8f;

__device__ __forceinline__ unsigned short f2bf(float f) {
    union { float f; unsigned u; } v; v.f = f;
    unsigned r = (v.u >> 16) & 1u;            // round to nearest even
    v.u += 0x7FFFu + r;
    return (unsigned short)(v.u >> 16);
}

__device__ __forceinline__ float sigf(float x) {
    return 1.0f / (1.0f + __expf(-x));
}

// Fragment load: uniform base (SGPR pair) + 32-bit per-lane element offset
// -> GVS addressing, two contiguous 16B chunks per lane.
__device__ __forceinline__ v16bf load_frag(const unsigned short* __restrict__ base, unsigned eo) {
    v8bf lo = *(const v8bf*)(const void*)(base + eo);        // K = kb + 0..7
    v8bf hi = *(const v8bf*)(const void*)(base + eo + 16);   // K = kb + 16..23
    return __builtin_shufflevector(lo, hi, 0,1,2,3,4,5,6,7,8,9,10,11,12,13,14,15);
}

#define WMMA_BF16(acc, a, b) \
    acc = __builtin_amdgcn_wmma_f32_16x16x32_bf16(false, (a), false, (b), (short)0, (acc), false, false)

// Monotonic grid-wide barrier (counter zeroed by prep kernel each launch)
__device__ __forceinline__ void grid_barrier(unsigned int* cnt, unsigned int expected) {
    __threadfence();
    __syncthreads();
    if (threadIdx.x == 0) {
        __hip_atomic_fetch_add(cnt, 1u, __ATOMIC_RELEASE, __HIP_MEMORY_SCOPE_AGENT);
        while (__hip_atomic_load(cnt, __ATOMIC_ACQUIRE, __HIP_MEMORY_SCOPE_AGENT) < expected) {
            __builtin_amdgcn_s_sleep(1);
        }
    }
    __syncthreads();
    __threadfence();
}

// ---------------------------------------------------------------------------
// Prep: pack weights [Wih | Whh] along K as bf16, combine biases, zero barrier
// ---------------------------------------------------------------------------
__global__ void k_prep_weights(const float* __restrict__ Wih0, const float* __restrict__ Whh0,
                               const float* __restrict__ bih0, const float* __restrict__ bhh0,
                               const float* __restrict__ Wih1, const float* __restrict__ Whh1,
                               const float* __restrict__ bih1, const float* __restrict__ bhh1,
                               unsigned short* __restrict__ W0, unsigned short* __restrict__ W1,
                               float* __restrict__ bias0, float* __restrict__ bias1,
                               unsigned int* __restrict__ cnt) {
    const size_t gid    = (size_t)blockIdx.x * blockDim.x + threadIdx.x;
    const size_t stride = (size_t)gridDim.x * blockDim.x;
    if (gid == 0) *cnt = 0u;
    for (size_t i = gid; i < (size_t)H3 * K0; i += stride) {
        size_t n = i / K0, k = i % K0;
        float v = (k < IN) ? Wih0[n * IN + k] : Whh0[n * H + (k - IN)];
        W0[i] = f2bf(v);
    }
    for (size_t i = gid; i < (size_t)H3 * K1; i += stride) {
        size_t n = i >> 11, k = i & (K1 - 1);
        float v = (k < H) ? Wih1[(n << 10) + k] : Whh1[(n << 10) + (k - H)];
        W1[i] = f2bf(v);
    }
    for (size_t h = gid; h < (size_t)H; h += stride) {
        bias0[0 * H + h] = bih0[0 * H + h] + bhh0[0 * H + h]; // r (sum ok)
        bias0[1 * H + h] = bih0[1 * H + h] + bhh0[1 * H + h]; // z (sum ok)
        bias0[2 * H + h] = bih0[2 * H + h];                   // inn
        bias0[3 * H + h] = bhh0[2 * H + h];                   // hn
        bias1[0 * H + h] = bih1[0 * H + h] + bhh1[0 * H + h];
        bias1[1 * H + h] = bih1[1 * H + h] + bhh1[1 * H + h];
        bias1[2 * H + h] = bih1[2 * H + h];
        bias1[3 * H + h] = bhh1[2 * H + h];
    }
}

__global__ void k_prep_x(const float* __restrict__ x, unsigned short* __restrict__ xb, size_t n) {
    const size_t gid    = (size_t)blockIdx.x * blockDim.x + threadIdx.x;
    const size_t stride = (size_t)gridDim.x * blockDim.x;
    for (size_t i = gid; i < n; i += stride) xb[i] = f2bf(x[i]);
}

__global__ void k_init_h(const float* __restrict__ ench,
                         float* __restrict__ h0f, float* __restrict__ h1f,
                         unsigned short* __restrict__ h0b, unsigned short* __restrict__ h1b) {
    int i = blockIdx.x * blockDim.x + threadIdx.x;
    if (i < B * H) {
        float v = ench[i];
        unsigned short bv = f2bf(v);
        h0f[i] = v;  h1f[i] = v;   // buffer index 0 (t=0 reads cur=0)
        h0b[i] = bv; h1b[i] = bv;
    }
}

// ---------------------------------------------------------------------------
// Persistent GRU kernel: 512 steps, 2 grid barriers/step.
// Wave w owns output tile (mt = w>>6 of 4, ct = w&63 of 64) -> 256 waves.
// Per tile: 4 f32 accumulators (r fused-K, z fused-K, inn input-K, hn hidden-K).
// K-loops are split into input/hidden segments; all hot accesses use
// uniform SGPR base + 32-bit per-lane offset (GVS addressing).
// ---------------------------------------------------------------------------
__global__ void __launch_bounds__(NTHR) k_gru_persistent(
    const unsigned short* __restrict__ xb,     // [B][T][IN] bf16
    const unsigned short* __restrict__ W0,     // [3H][K0]   bf16
    const unsigned short* __restrict__ W1,     // [3H][K1]   bf16
    const float* __restrict__ bias0,           // [4][H]
    const float* __restrict__ bias1,           // [4][H]
    float* __restrict__ h0f, float* __restrict__ h1f,                   // [2][B][H]
    unsigned short* __restrict__ h0b, unsigned short* __restrict__ h1b, // [2][B][H]
    float* __restrict__ out,                   // [B][T][H] + tail [B][H]
    unsigned int* __restrict__ cnt) {

    const int lane  = threadIdx.x & 31;
    const int wid   = blockIdx.x * (NTHR / 32) + (threadIdx.x >> 5); // 0..255
    const int mt    = wid >> 6;                 // 0..3   (16 rows of B)
    const int ct    = wid & 63;                 // 0..63  (16 cols of H)
    const int nlo   = lane & 15;
    const int khalf = (lane >> 4) << 3;         // 0 or 8 (A/B K-offset per lane)
    const int mrow  = mt * 16 + nlo;            // A row this lane loads
    const int crow  = mt * 16 + ((lane >> 4) << 3); // C base row (add r=0..7)
    const int hcol  = ct * 16 + nlo;            // C column this lane holds

    const float b0r = bias0[0 * H + hcol], b0z = bias0[1 * H + hcol];
    const float b0i = bias0[2 * H + hcol], b0h = bias0[3 * H + hcol];
    const float b1r = bias1[0 * H + hcol], b1z = bias1[1 * H + hcol];
    const float b1i = bias1[2 * H + hcol], b1h = bias1[3 * H + hcol];

    // 32-bit per-lane element offsets (uniform bases stay in SGPRs)
    const unsigned w0r_o = (unsigned)((0 * H + ct * 16 + nlo) * K0 + khalf);
    const unsigned w0z_o = (unsigned)((1 * H + ct * 16 + nlo) * K0 + khalf);
    const unsigned w0n_o = (unsigned)((2 * H + ct * 16 + nlo) * K0 + khalf);
    const unsigned w1r_o = (unsigned)((0 * H + ct * 16 + nlo) * K1 + khalf);
    const unsigned w1z_o = (unsigned)((1 * H + ct * 16 + nlo) * K1 + khalf);
    const unsigned w1n_o = (unsigned)((2 * H + ct * 16 + nlo) * K1 + khalf);
    const unsigned xrow_o0 = (unsigned)(mrow * T * IN + khalf);   // + t*IN per step
    const unsigned hrow_o  = (unsigned)(mrow * H + khalf);        // + cur*B*H per step

    unsigned int bars = 0;

    for (int t = 0; t < T; ++t) {
        const unsigned cur = (unsigned)(t & 1), nxt = cur ^ 1u;
        const unsigned curBH = cur * (unsigned)(B * H);
        const unsigned nxtBH = nxt * (unsigned)(B * H);

        { // ---------------- layer 0: h0_new = GRU(x_t, h0_cur) ----------------
            v8f cr = {0,0,0,0,0,0,0,0}, cz = {0,0,0,0,0,0,0,0};
            v8f ci = {0,0,0,0,0,0,0,0}, ch = {0,0,0,0,0,0,0,0};
            const unsigned xo = xrow_o0 + (unsigned)t * IN;
            const unsigned ho = curBH + hrow_o;

            // input segment: A = x_t, K = [0, IN)
            #pragma unroll 4
            for (unsigned k0 = 0; k0 < IN; k0 += 32) {
                v16bf a = load_frag(xb, xo + k0);
                WMMA_BF16(cr, a, load_frag(W0, w0r_o + k0));
                WMMA_BF16(cz, a, load_frag(W0, w0z_o + k0));
                WMMA_BF16(ci, a, load_frag(W0, w0n_o + k0));
            }
            // hidden segment: A = h0_cur, K = [IN, IN+H)
            #pragma unroll 4
            for (unsigned k0 = 0; k0 < H; k0 += 32) {
                v16bf a = load_frag(h0b, ho + k0);
                WMMA_BF16(cr, a, load_frag(W0, w0r_o + IN + k0));
                WMMA_BF16(cz, a, load_frag(W0, w0z_o + IN + k0));
                WMMA_BF16(ch, a, load_frag(W0, w0n_o + IN + k0));
            }

            #pragma unroll
            for (int r = 0; r < 8; ++r) {
                const unsigned e = (unsigned)(crow + r) * H + (unsigned)hcol;
                float rg = sigf(cr[r] + b0r);
                float zg = sigf(cz[r] + b0z);
                float ng = tanhf(ci[r] + b0i + rg * (ch[r] + b0h));
                float hp = h0f[curBH + e];
                float hv = (1.0f - zg) * ng + zg * hp;
                h0f[nxtBH + e] = hv;
                h0b[nxtBH + e] = f2bf(hv);
            }
        }
        bars++; grid_barrier(cnt, bars * NBLK);

        { // ---------------- layer 1: h1_new = GRU(h0_new, h1_cur) --------------
            v8f cr = {0,0,0,0,0,0,0,0}, cz = {0,0,0,0,0,0,0,0};
            v8f ci = {0,0,0,0,0,0,0,0}, ch = {0,0,0,0,0,0,0,0};
            const unsigned a0o = nxtBH + hrow_o; // h0_new
            const unsigned a1o = curBH + hrow_o; // h1_cur

            // input segment: A = h0_new, K = [0, H)
            #pragma unroll 4
            for (unsigned k0 = 0; k0 < H; k0 += 32) {
                v16bf a = load_frag(h0b, a0o + k0);
                WMMA_BF16(cr, a, load_frag(W1, w1r_o + k0));
                WMMA_BF16(cz, a, load_frag(W1, w1z_o + k0));
                WMMA_BF16(ci, a, load_frag(W1, w1n_o + k0));
            }
            // hidden segment: A = h1_cur, K = [H, 2H)
            #pragma unroll 4
            for (unsigned k0 = 0; k0 < H; k0 += 32) {
                v16bf a = load_frag(h1b, a1o + k0);
                WMMA_BF16(cr, a, load_frag(W1, w1r_o + H + k0));
                WMMA_BF16(cz, a, load_frag(W1, w1z_o + H + k0));
                WMMA_BF16(ch, a, load_frag(W1, w1n_o + H + k0));
            }

            #pragma unroll
            for (int r = 0; r < 8; ++r) {
                const unsigned e = (unsigned)(crow + r) * H + (unsigned)hcol;
                float rg = sigf(cr[r] + b1r);
                float zg = sigf(cz[r] + b1z);
                float ng = tanhf(ci[r] + b1i + rg * (ch[r] + b1h));
                float hp = h1f[curBH + e];
                float hv = (1.0f - zg) * ng + zg * hp;
                h1f[nxtBH + e] = hv;
                h1b[nxtBH + e] = f2bf(hv);
                out[(unsigned)(crow + r) * (T * H) + (unsigned)t * H + (unsigned)hcol] = hv;
            }
        }
        bars++; grid_barrier(cnt, bars * NBLK);
    }

    // final carry (layer-1 hidden) -> tail of d_out; final state lives in buffer T&1
    const float* hfin = h1f + (size_t)(T & 1) * B * H;
    float* tail = out + (size_t)B * T * H;
    for (int i = blockIdx.x * NTHR + threadIdx.x; i < B * H; i += NBLK * NTHR)
        tail[i] = hfin[i];
}

// ---------------------------------------------------------------------------
extern "C" void kernel_launch(void* const* d_in, const int* in_sizes, int n_in,
                              void* d_out, int out_size, void* d_ws, size_t ws_size,
                              hipStream_t stream) {
    (void)in_sizes; (void)n_in; (void)out_size; (void)ws_size;
    const float* x    = (const float*)d_in[0];
    // d_in[1] = encoder_out (unused by reference config)
    const float* ench = (const float*)d_in[2];
    const float* Wih0 = (const float*)d_in[3];
    const float* Whh0 = (const float*)d_in[4];
    const float* bih0 = (const float*)d_in[5];
    const float* bhh0 = (const float*)d_in[6];
    const float* Wih1 = (const float*)d_in[7];
    const float* Whh1 = (const float*)d_in[8];
    const float* bih1 = (const float*)d_in[9];
    const float* bhh1 = (const float*)d_in[10];
    float* out = (float*)d_out;

    char* p = (char*)d_ws;
    auto carve = [&](size_t bytes) -> char* {
        char* r = p; p += (bytes + 255) & ~(size_t)255; return r;
    };
    unsigned int*   cnt   = (unsigned int*)  carve(256);
    unsigned short* xb    = (unsigned short*)carve((size_t)B * T * IN * 2);
    unsigned short* W0    = (unsigned short*)carve((size_t)H3 * K0 * 2);
    unsigned short* W1    = (unsigned short*)carve((size_t)H3 * K1 * 2);
    float*          bias0 = (float*)         carve((size_t)4 * H * 4);
    float*          bias1 = (float*)         carve((size_t)4 * H * 4);
    float*          h0f   = (float*)         carve((size_t)2 * B * H * 4);
    float*          h1f   = (float*)         carve((size_t)2 * B * H * 4);
    unsigned short* h0b   = (unsigned short*)carve((size_t)2 * B * H * 2);
    unsigned short* h1b   = (unsigned short*)carve((size_t)2 * B * H * 2);

    k_prep_weights<<<2048, 256, 0, stream>>>(Wih0, Whh0, bih0, bhh0,
                                             Wih1, Whh1, bih1, bhh1,
                                             W0, W1, bias0, bias1, cnt);
    k_prep_x<<<4096, 256, 0, stream>>>(x, xb, (size_t)B * T * IN);
    k_init_h<<<(B * H + 255) / 256, 256, 0, stream>>>(ench, h0f, h1f, h0b, h1b);
    k_gru_persistent<<<NBLK, NTHR, 0, stream>>>(xb, W0, W1, bias0, bias1,
                                                h0f, h1f, h0b, h1b, out, cnt);
}
